// MedicalWindowAttention_60387240181900
// MI455X (gfx1250) — matrix operand order repaired and checked by reading, not compiled, not run on hardware
//
#include <hip/hip_runtime.h>
#include <hip/hip_bf16.h>

typedef __bf16 bf16_t;
typedef __attribute__((ext_vector_type(16))) __bf16 v16bf;
typedef __attribute__((ext_vector_type(8)))  __bf16 v8bf;
typedef __attribute__((ext_vector_type(8)))  float  v8f;

#define DIMC   384
#define HEADS  12
#define HD     32
#define NTOK   49
#define NPAD   64
#define NWIN   64
#define QSCALE 0.17677669529663687f   // 32^-0.5

union pk4 { bf16_t h[4]; uint2 u; };

__device__ __forceinline__ v8f zero8() {
  v8f z;
#pragma unroll
  for (int i = 0; i < 8; ++i) z[i] = 0.f;
  return z;
}

__device__ __forceinline__ v8f wmma_bf16(v16bf a, v16bf b, v8f c) {
  // v_wmma_f32_16x16x32_bf16  D = A(16x32) * B(32x16) + C(16x16 f32)
  return __builtin_amdgcn_wmma_f32_16x16x32_bf16(false, a, false, b, (short)0, c, false, false);
}

// A fragment (16x32 bf16) from row-major LDS [row][k], stride in elements.
// Lane L holds row row0+(L&15); elems e=0..15 -> k0+off+e, off=(L>=16)*16.
__device__ __forceinline__ v16bf a_frag_lds(const bf16_t* base, int stride, int row0, int k0) {
  int lane = threadIdx.x & 31;
  const bf16_t* p = base + (row0 + (lane & 15)) * stride + k0 + ((lane >> 4) << 4);
  v8bf lo = *(const v8bf*)p;
  v8bf hi = *(const v8bf*)(p + 8);
  v16bf r;
#pragma unroll
  for (int e = 0; e < 8; ++e) { r[e] = lo[e]; r[8 + e] = hi[e]; }
  return r;
}

// B fragment (32x16 bf16): lane L holds column col0+(L&15); elems e=0..15 -> k = k0 + (L>=16)*16 + e.
// Source is row-major [col][k] (i.e. "transposed" storage), so reads are 32B contiguous.
__device__ __forceinline__ v16bf b_frag_lds(const bf16_t* base, int stride, int col0, int k0) {
  int lane = threadIdx.x & 31;
  const bf16_t* p = base + (col0 + (lane & 15)) * stride + k0 + ((lane >> 4) << 4);
  return *(const v16bf*)p;
}

// B fragment from global fp32 weight W[out][in] row-major (out features = B columns).
__device__ __forceinline__ v16bf b_frag_w(const float* W, int col0, int k0) {
  int lane = threadIdx.x & 31;
  const float4* p4 = (const float4*)(W + (size_t)(col0 + (lane & 15)) * DIMC + k0 + ((lane >> 4) << 4));
  float4 f0 = p4[0], f1 = p4[1], f2 = p4[2], f3 = p4[3];
  v16bf r;
  r[0]  = (bf16_t)f0.x; r[1]  = (bf16_t)f0.y; r[2]  = (bf16_t)f0.z; r[3]  = (bf16_t)f0.w;
  r[4]  = (bf16_t)f1.x; r[5]  = (bf16_t)f1.y; r[6]  = (bf16_t)f1.z; r[7]  = (bf16_t)f1.w;
  r[8]  = (bf16_t)f2.x; r[9]  = (bf16_t)f2.y; r[10] = (bf16_t)f2.z; r[11] = (bf16_t)f2.w;
  r[12] = (bf16_t)f3.x; r[13] = (bf16_t)f3.y; r[14] = (bf16_t)f3.z; r[15] = (bf16_t)f3.w;
  return r;
}

__device__ __forceinline__ int rel_index(int n, int m) {
  int dh = n / 7 - m / 7 + 6;
  int dw = n % 7 - m % 7 + 6;
  return dh * 13 + dw;   // index into [169, HEADS] table
}

__global__ __launch_bounds__(256, 1)
void mwa_fused(const float* __restrict__ x,     const float* __restrict__ mask,
               const float* __restrict__ qkvw,  const float* __restrict__ qkvb,
               const float* __restrict__ relb,  const float* __restrict__ cscale,
               const float* __restrict__ q1w,   const float* __restrict__ q1b,
               const float* __restrict__ q2w,   const float* __restrict__ q2b,
               const float* __restrict__ projw, const float* __restrict__ projb,
               float* __restrict__ out)
{
  extern __shared__ char smem[];
  bf16_t* xs   = (bf16_t*)smem;            // [64][384]  x (bf16), later reused as attention output
  bf16_t* qs   = xs  + NPAD * DIMC;        // [64][384]  Q * QSCALE, later reused as gated output
  bf16_t* ks   = qs  + NPAD * DIMC;        // [64][384]  K
  bf16_t* vsT  = ks  + NPAD * DIMC;        // [384][64]  V transposed (col-major per feature)
  bf16_t* pbuf = vsT + DIMC * NPAD;        // 8 waves x [16][64] softmax P staging
  bf16_t* g1b  = pbuf + 8 * 16 * 64;       // [64][96]   relu(out @ q1_w^T + q1_b)
  float*  gate = (float*)(g1b + NPAD * 96);// [64]       sigmoid gate
  bf16_t* os   = xs;                       // reuse
  bf16_t* gb   = qs;                       // reuse

  const int b     = blockIdx.x;
  const int wi    = b & (NWIN - 1);        // window-mask index = b % 64
  const int wave  = threadIdx.x >> 5;
  const int lane  = threadIdx.x & 31;
  const int col   = lane & 15;
  const int rbase = (lane >> 4) << 3;      // row offset of this half-wave in C/D tiles

  // ---------- Phase 0: stage x window into LDS (bf16, packed 4-wide), zero-pad rows 49..63 ----------
  const float4* xw4 = (const float4*)(x + (size_t)b * (NTOK * DIMC));
  for (int i = threadIdx.x; i < (NTOK * DIMC) / 4; i += 256) {
    float4 f = xw4[i];
    pk4 p;
    p.h[0] = (bf16_t)f.x; p.h[1] = (bf16_t)f.y; p.h[2] = (bf16_t)f.z; p.h[3] = (bf16_t)f.w;
    *(uint2*)(xs + 4 * i) = p.u;
  }
  {
    uint2 z; z.x = 0u; z.y = 0u;
    for (int i = threadIdx.x; i < ((NPAD - NTOK) * DIMC) / 4; i += 256)
      *(uint2*)(xs + NTOK * DIMC + 4 * i) = z;
  }
  __syncthreads();

  // ---------- Phase 1: QKV = x @ qkv_w^T + qkv_b   ([64x384] x [384x1152]) ----------
  // 72 col-tiles of 16 -> 9 per wave; a 16-wide tile never straddles the q/k/v boundary
  // (24 tiles per segment), so the q/k/v split below is wave-uniform in ct.
  for (int ct = wave * 9; ct < wave * 9 + 9; ++ct) {
    v8f acc[4];
#pragma unroll
    for (int rt = 0; rt < 4; ++rt) acc[rt] = zero8();
    for (int kt = 0; kt < DIMC / 32; ++kt) {
      v16bf bw = b_frag_w(qkvw, ct * 16, kt * 32);
#pragma unroll
      for (int rt = 0; rt < 4; ++rt)
        acc[rt] = wmma_bf16(a_frag_lds(xs, DIMC, rt * 16, kt * 32), bw, acc[rt]);
    }
    const int   o   = ct * 16 + col;
    const float bia = qkvb[o];
    if (ct < 24) {                     // Q segment (pre-scaled)
      bf16_t* dst = qs + o;
#pragma unroll
      for (int rt = 0; rt < 4; ++rt)
#pragma unroll
        for (int r = 0; r < 8; ++r)
          dst[(rt * 16 + rbase + r) * DIMC] = (bf16_t)((acc[rt][r] + bia) * QSCALE);
    } else if (ct < 48) {              // K segment
      bf16_t* dst = ks + (o - DIMC);
#pragma unroll
      for (int rt = 0; rt < 4; ++rt)
#pragma unroll
        for (int r = 0; r < 8; ++r)
          dst[(rt * 16 + rbase + r) * DIMC] = (bf16_t)(acc[rt][r] + bia);
    } else {                           // V segment, stored transposed [feature][row]
      bf16_t* dst = vsT + (o - 2 * DIMC) * NPAD;
#pragma unroll
      for (int rt = 0; rt < 4; ++rt)
#pragma unroll
        for (int r = 0; r < 8; ++r)
          dst[rt * 16 + rbase + r] = (bf16_t)(acc[rt][r] + bia);
    }
  }
  __syncthreads();

  // ---------- Phase 2: per-head attention (heads striped over waves) ----------
  bf16_t* pw = pbuf + wave * 16 * 64;
  for (int h = wave; h < HEADS; h += 8) {
    const float cs = cscale[h];
    for (int rt = 0; rt < 4; ++rt) {
      // S tile row: [16 x 64] = Q_rt (16x32) x K^T, one WMMA per 16-col tile (K == HD == 32)
      v16bf qa = a_frag_lds(qs, DIMC, rt * 16, h * HD);
      v8f s[4];
#pragma unroll
      for (int ct = 0; ct < 4; ++ct)
        s[ct] = wmma_bf16(qa, b_frag_lds(ks, DIMC, ct * 16, h * HD), zero8());

      // bias + mask + contrast scale, then row softmax (rows live per-VGPR in 16-lane halves)
      float rsum[8];
#pragma unroll
      for (int r = 0; r < 8; ++r) {
        int   M = rt * 16 + rbase + r;
        float vals[4];
#pragma unroll
        for (int ct = 0; ct < 4; ++ct) {
          int   m = ct * 16 + col;
          float v = s[ct][r];
          if (m < NTOK) {
            if (M < NTOK)
              v += relb[rel_index(M, m) * HEADS + h] + mask[(size_t)wi * NTOK * NTOK + M * NTOK + m];
            v *= cs;
          } else v = -1e30f;
          vals[ct] = v;
        }
        float mx = fmaxf(fmaxf(vals[0], vals[1]), fmaxf(vals[2], vals[3]));
#pragma unroll
        for (int off = 8; off >= 1; off >>= 1) mx = fmaxf(mx, __shfl_xor(mx, off, 32));
        float sum = 0.f;
#pragma unroll
        for (int ct = 0; ct < 4; ++ct) { float p = __expf(vals[ct] - mx); vals[ct] = p; sum += p; }
#pragma unroll
        for (int off = 8; off >= 1; off >>= 1) sum += __shfl_xor(sum, off, 32);
        rsum[r] = sum;
#pragma unroll
        for (int ct = 0; ct < 4; ++ct)
          pw[(rbase + r) * 64 + ct * 16 + col] = (bf16_t)vals[ct];
      }
      // same-wave LDS RAW (cross-lane): wait all DS ops before re-reading pw as A fragments
      asm volatile("s_wait_dscnt 0x0" ::: "memory");

      // O tile = P (16x64) x V_h (64x32): 2 K-chunks x 2 col-tiles
      v8f oa[2] = { zero8(), zero8() };
#pragma unroll
      for (int kt = 0; kt < 2; ++kt) {
        v16bf pa = a_frag_lds(pw, 64, 0, kt * 32);
#pragma unroll
        for (int c2 = 0; c2 < 2; ++c2)
          oa[c2] = wmma_bf16(pa, b_frag_lds(vsT, NPAD, h * HD + c2 * 16, kt * 32), oa[c2]);
      }
#pragma unroll
      for (int c2 = 0; c2 < 2; ++c2)
#pragma unroll
        for (int r = 0; r < 8; ++r) {
          int row = rt * 16 + rbase + r;
          os[row * DIMC + h * HD + c2 * 16 + col] = (bf16_t)(oa[c2][r] / rsum[r]);
        }
    }
  }
  __syncthreads();

  // ---------- Phase 3a: g1 = relu(out @ q1_w^T + q1_b)  ([64x384] x [384x96]) ----------
  for (int t = wave * 3; t < wave * 3 + 3; ++t) {
    int rt = t / 6, ct = t % 6;
    v8f acc = zero8();
    for (int kt = 0; kt < DIMC / 32; ++kt)
      acc = wmma_bf16(a_frag_lds(os, DIMC, rt * 16, kt * 32), b_frag_w(q1w, ct * 16, kt * 32), acc);
    int   o   = ct * 16 + col;
    float bia = q1b[o];
#pragma unroll
    for (int r = 0; r < 8; ++r)
      g1b[(rt * 16 + rbase + r) * 96 + o] = (bf16_t)fmaxf(acc[r] + bia, 0.f);
  }
  __syncthreads();

  // ---------- Phase 3b: gate = sigmoid(g1 @ q2_w^T + q2_b)  (4 lanes per row + quad reduce) ----------
  {
    int   row = threadIdx.x >> 2, sub = threadIdx.x & 3;
    float a   = 0.f;
    for (int c = sub * 24; c < sub * 24 + 24; ++c)
      a += (float)g1b[row * 96 + c] * q2w[c];
    a += __shfl_xor(a, 1, 32);
    a += __shfl_xor(a, 2, 32);
    if (sub == 0) gate[row] = 1.f / (1.f + __expf(-(a + q2b[0])));
  }
  __syncthreads();

  // ---------- Phase 3c: gated = out * gate (4-wide packed) ----------
  for (int i = threadIdx.x; i < (NPAD * DIMC) / 4; i += 256) {
    float g = gate[(4 * i) / DIMC];
    pk4 in, outp;
    in.u = *(const uint2*)(os + 4 * i);
#pragma unroll
    for (int e = 0; e < 4; ++e) outp.h[e] = (bf16_t)((float)in.h[e] * g);
    *(uint2*)(gb + 4 * i) = outp.u;
  }
  __syncthreads();

  // ---------- Phase 3d: final = gated @ proj_w^T + proj_b  ([64x384] x [384x384]) ----------
  float* ow = out + (size_t)b * (NTOK * DIMC);
  for (int t = wave * 12; t < wave * 12 + 12; ++t) {
    int rt = t / 24, ct = t % 24;
    v8f acc = zero8();
    for (int kt = 0; kt < DIMC / 32; ++kt)
      acc = wmma_bf16(a_frag_lds(gb, DIMC, rt * 16, kt * 32), b_frag_w(projw, ct * 16, kt * 32), acc);
    int   o   = ct * 16 + col;
    float bia = projb[o];
#pragma unroll
    for (int r = 0; r < 8; ++r) {
      int row = rt * 16 + rbase + r;
      if (row < NTOK) ow[row * DIMC + o] = acc[r] + bia;
    }
  }
}

extern "C" void kernel_launch(void* const* d_in, const int* in_sizes, int n_in,
                              void* d_out, int out_size, void* d_ws, size_t ws_size,
                              hipStream_t stream) {
  (void)n_in; (void)out_size; (void)d_ws; (void)ws_size;
  const float* x      = (const float*)d_in[0];
  const float* mask   = (const float*)d_in[1];
  // d_in[2] = quality_scores: only gates the branch in the reference, values unused
  const float* qkvw   = (const float*)d_in[3];
  const float* qkvb   = (const float*)d_in[4];
  const float* relb   = (const float*)d_in[5];
  const float* cs     = (const float*)d_in[6];
  const float* q1w    = (const float*)d_in[7];
  const float* q1b    = (const float*)d_in[8];
  const float* q2w    = (const float*)d_in[9];
  const float* q2b    = (const float*)d_in[10];
  const float* projw  = (const float*)d_in[11];
  const float* projb  = (const float*)d_in[12];
  float* out = (float*)d_out;

  const int B_ = in_sizes[0] / (NTOK * DIMC);   // 4096 windows

  // dynamic LDS: 4 * 64*384*2 + 8*16*64*2 + 64*96*2 + 64*4 = 225,536 bytes
  const size_t shmem = (size_t)(4 * NPAD * DIMC + 8 * 16 * 64 + NPAD * 96) * sizeof(bf16_t)
                     + NPAD * sizeof(float);
  hipFuncSetAttribute(reinterpret_cast<const void*>(mwa_fused),
                      hipFuncAttributeMaxDynamicSharedMemorySize, (int)shmem);

  mwa_fused<<<dim3(B_), dim3(256), shmem, stream>>>(
      x, mask, qkvw, qkvb, relb, cs, q1w, q1b, q2w, q2b, projw, projb, out);
}
